// FocusAttention_16149077033367
// MI455X (gfx1250) — matrix operand order repaired and checked by reading, not compiled, not run on hardware
//
#include <hip/hip_runtime.h>
#include <hip/hip_bf16.h>
#include <math.h>

// ---------------- problem constants ----------------
#define B_    4
#define NA_   256
#define NV_   2048
#define DIM_  1024
#define H_    16
#define DH_   64
#define TOPK_ 64
#define ROWS_A (B_*NA_)   // 1024
#define ROWS_V (B_*NV_)   // 8192

typedef __bf16 bf16_t;
typedef __attribute__((ext_vector_type(16))) __bf16 v16bf;
typedef __attribute__((ext_vector_type(8)))  float  v8f;

__device__ __forceinline__ v8f wmma_bf16(v16bf a, v16bf b, v8f c) {
  return __builtin_amdgcn_wmma_f32_16x16x32_bf16(false, a, false, b, (short)0, c, false, false);
}

// fast hardware reciprocal / rsqrt (v_rcp_f32 / v_rsq_f32)
__device__ __forceinline__ float fast_rcp(float x)  { return __builtin_amdgcn_rcpf(x); }
__device__ __forceinline__ float fast_rsq(float x)  { return __builtin_amdgcn_rsqf(x); }
__device__ __forceinline__ float fast_sigmoid(float x) {
  return fast_rcp(1.f + __expf(-x));
}

// monotonic float->uint mapping (order-preserving, handles -inf)
__device__ __forceinline__ unsigned f2mono(float f) {
  unsigned u = __float_as_uint(f);
  return (u & 0x80000000u) ? ~u : (u | 0x80000000u);
}

// CDNA5 async global->LDS copy (ASYNCcnt-tracked), 16 bytes per lane.
// VDST = VGPR holding LDS byte offset; VADDR = 64-bit global address.
__device__ __forceinline__ void async_load_b128(const void* gptr, void* lptr) {
  unsigned lds = (unsigned)(size_t)lptr;            // LDS aperture: addr[31:0] = LDS offset
  unsigned long long ga = (unsigned long long)gptr;
  asm volatile("global_load_async_to_lds_b128 %0, %1, off"
               :: "v"(lds), "v"(ga) : "memory");
}
__device__ __forceinline__ void wait_async0() {
  asm volatile("s_wait_asynccnt 0x0" ::: "memory");
}

// ---------------- kernel: zero init ----------------
__global__ void zero_kernel(float* __restrict__ p, int n) {
  int i = blockIdx.x * 256 + threadIdx.x;
  if (i < n) p[i] = 0.f;
}

// ---------------- kernel: layernorm (optionally two affine outputs) ----------------
__global__ __launch_bounds__(256)
void ln_dual_kernel(const float* __restrict__ x,
                    const float* __restrict__ g1, const float* __restrict__ b1,
                    const float* __restrict__ g2, const float* __restrict__ b2,
                    bf16_t* __restrict__ o1, bf16_t* __restrict__ o2) {
  const int row = blockIdx.x;
  const int tid = threadIdx.x;
  const float* xr = x + (size_t)row * DIM_;

  float4 lx = *(const float4*)(xr + tid * 4);
  float s  = lx.x + lx.y + lx.z + lx.w;
  float sq = lx.x*lx.x + lx.y*lx.y + lx.z*lx.z + lx.w*lx.w;

  __shared__ float shs[8], shq[8];
  for (int o = 16; o; o >>= 1) { s += __shfl_xor(s, o); sq += __shfl_xor(sq, o); }
  if ((tid & 31) == 0) { shs[tid >> 5] = s; shq[tid >> 5] = sq; }
  __syncthreads();
  if (tid == 0) {
    float ts = 0.f, tq = 0.f;
    for (int i = 0; i < 8; ++i) { ts += shs[i]; tq += shq[i]; }
    shs[0] = ts; shq[0] = tq;
  }
  __syncthreads();
  const float mean = shs[0] * (1.0f / DIM_);
  const float var  = shq[0] * (1.0f / DIM_) - mean * mean;
  const float rstd = fast_rsq(var + 1e-5f);

  float v[4] = { lx.x, lx.y, lx.z, lx.w };
  #pragma unroll
  for (int i = 0; i < 4; ++i) {
    int c = tid * 4 + i;
    float xn = (v[i] - mean) * rstd;
    o1[(size_t)row * DIM_ + c] = (bf16_t)(xn * g1[c] + b1[c]);
    if (o2) o2[(size_t)row * DIM_ + c] = (bf16_t)(xn * g2[c] + b2[c]);
  }
}

// ---------------- kernel: WMMA GEMM  out = X @ W^T + bias ----------------
// X: [M, K] bf16 row-major; W: [N, K] fp32 row-major; K = N = DIM_
// mode 0: out_bf = bf16(result)
// mode 1: out_f = sigmoid(result); accumulate gate_mean
// mode 2: out2_f = result; out_f = result * gmat
#define BM 64
#define BN 64
#define BK 32
__global__ __launch_bounds__(128)
void gemm_wmma_kernel(const bf16_t* __restrict__ X, const float* __restrict__ W,
                      const float* __restrict__ bias,
                      bf16_t* __restrict__ out_bf, float* __restrict__ out_f,
                      const float* __restrict__ gmat, float* __restrict__ out2_f,
                      float* __restrict__ gate_mean_slot, int M, int mode) {
  constexpr int N = DIM_, K = DIM_;
  __shared__ __align__(16) bf16_t Xs[BM][BK];
  __shared__ __align__(16) bf16_t Ws[BN][BK];

  const int m0 = blockIdx.x * BM;
  const int n0 = blockIdx.y * BN;
  const int tid = threadIdx.x;
  const int lane = tid & 31, wv = tid >> 5;
  const int ar = lane & 15;
  const int kb = (lane >> 4) * 8;
  const int kb2 = (lane >> 4) * 16;

  v8f acc[4] = {};

  for (int k0 = 0; k0 < K; k0 += BK) {
    // stage X tile via CDNA5 async global->LDS (bf16, 2x16B per thread)
    {
      int r = tid >> 1, half = tid & 1;
      const bf16_t* src = X + (size_t)(m0 + r) * K + k0 + half * 16;
      async_load_b128(src,     &Xs[r][half * 16]);
      async_load_b128(src + 8, &Xs[r][half * 16 + 8]);
    }
    // stage W tile, fp32 -> bf16 (needs VALU convert, so VGPR path)
    {
      int r = tid >> 1, half = tid & 1;
      const float* src = W + (size_t)(n0 + r) * K + k0 + half * 16;
      bf16_t* dst = &Ws[r][half * 16];
      #pragma unroll
      for (int j = 0; j < 16; ++j) dst[j] = (bf16_t)src[j];
    }
    wait_async0();
    __syncthreads();

    v16bf afrag;
    {
      int row = wv * 16 + ar;
      #pragma unroll
      for (int j = 0; j < 8; ++j) { afrag[j] = Xs[row][kb2 + j]; afrag[8 + j] = Xs[row][kb2 + 8 + j]; }
    }
    #pragma unroll
    for (int nt = 0; nt < 4; ++nt) {
      int nrow = nt * 16 + ar;
      v16bf bfrag;
      #pragma unroll
      for (int e = 0; e < 16; ++e) bfrag[e] = Ws[nrow][kb2 + e];
      acc[nt] = wmma_bf16(afrag, bfrag, acc[nt]);
    }
    __syncthreads();
  }

  // epilogue (fully unrolled: no dynamic VGPR indexing)
  const int rr = (lane >> 4) * 8, cc = lane & 15;
  float gsum = 0.f;
  #pragma unroll
  for (int nt = 0; nt < 4; ++nt) {
    int col = n0 + nt * 16 + cc;
    float bv = bias[col];
    #pragma unroll
    for (int i = 0; i < 8; ++i) {
      int row = m0 + wv * 16 + rr + i;
      float v = acc[nt][i] + bv;
      size_t idx = (size_t)row * N + col;
      if (mode == 0) {
        out_bf[idx] = (bf16_t)v;
      } else if (mode == 1) {
        float g = fast_sigmoid(v);
        out_f[idx] = g;
        gsum += g;
      } else {
        out2_f[idx] = v;
        out_f[idx]  = v * gmat[idx];
      }
    }
  }
  if (mode == 1) {
    for (int o = 16; o; o >>= 1) gsum += __shfl_xor(gsum, o);
    if (lane == 0) atomicAdd(gate_mean_slot, gsum * (1.0f / ((float)M * (float)N)));
  }
}

// ---------------- kernel: attention core ----------------
// one block per (b, h, 16-row q tile); 128 threads = 4 waves
// scalars: [0]=attn_max [1]=ent_norm [2]=ent_abs [3]=gate_mean
__global__ __launch_bounds__(128)
void attention_kernel(const bf16_t* __restrict__ Q, const bf16_t* __restrict__ Km,
                      const bf16_t* __restrict__ Vm, const unsigned char* __restrict__ vmask,
                      bf16_t* __restrict__ hvcat, float* __restrict__ attn_avg,
                      float* __restrict__ scalars) {
  __shared__ __align__(16) float  sc[16][NV_ + 4];  // 16 x 2052 fp32 = 128.25 KB (CDNA5 LDS)
  __shared__ __align__(16) bf16_t qs[16][DH_];
  __shared__ __align__(16) bf16_t Vs[32][72];       // 32-key Val chunk, +8 pad (16B-aligned rows)

  const int blk = blockIdx.x;
  const int qt = blk & 15;
  const int h  = (blk >> 4) & 15;
  const int b  = blk >> 8;
  const int q0 = qt * 16;
  const int tid = threadIdx.x, lane = tid & 31, wv = tid >> 5;
  const int ar = lane & 15;
  const int kb = (lane >> 4) * 8;
  const int kb2 = (lane >> 4) * 16;

  // stage Q tile into LDS
  for (int i = tid; i < 16 * DH_; i += 128) {
    int r = i >> 6, d = i & 63;
    qs[r][d] = Q[(size_t)(b * NA_ + q0 + r) * DIM_ + h * DH_ + d];
  }
  __syncthreads();

  v16bf aq0, aq1;
  #pragma unroll
  for (int j = 0; j < 8; ++j) {
    aq0[j] = qs[ar][kb2 + j];      aq0[8 + j] = qs[ar][kb2 + 8 + j];
    aq1[j] = qs[ar][32 + kb2 + j]; aq1[8 + j] = qs[ar][40 + kb2 + j];
  }

  // ---- scores: Q (16x64) x K^T (64x16) per key tile, 2 chained WMMAs ----
  const float scale = 0.125f;  // 1/sqrt(64)
  for (int kt = wv; kt < NV_ / 16; kt += 4) {
    int keyr = kt * 16 + ar;
    const bf16_t* kr = Km + (size_t)(b * NV_ + keyr) * DIM_ + h * DH_;
    if (kt + 4 < NV_ / 16)  // prefetch next K tile for this wave
      __builtin_prefetch(kr + (size_t)64 * DIM_, 0, 1);
    v16bf bk0, bk1;
    #pragma unroll
    for (int e = 0; e < 16; ++e) { bk0[e] = kr[kb2 + e]; bk1[e] = kr[32 + kb2 + e]; }
    v8f acc = {};
    acc = wmma_bf16(aq0, bk0, acc);
    acc = wmma_bf16(aq1, bk1, acc);
    const int rr = (lane >> 4) * 8, cc = lane & 15;
    const int keyc = kt * 16 + cc;
    const bool ok = vmask[b * NV_ + keyc] != 0;
    #pragma unroll
    for (int i = 0; i < 8; ++i) {
      float s = acc[i] * scale;
      sc[rr + i][keyc] = ok ? s : -__builtin_inff();
    }
  }
  __syncthreads();

  // ---- per-row top-64 (binary radix select) + softmax + entropy ----
  const float CLIP = 1e-8f, NLCLIP = 18.420680743952367f;  // -ln(1e-8)
  const float LOGK = 4.1588830833596715f;                  // ln(64)
  for (int rsel = 0; rsel < 4; ++rsel) {
    int row = wv * 4 + rsel;
    float* rs = sc[row];

    unsigned prefix = 0u; int kneed = TOPK_;
    for (int bit = 31; bit >= 0; --bit) {
      unsigned cand = prefix | (1u << bit);
      int cnt = 0;
      for (int t = lane; t < NV_; t += 32) {
        unsigned u = f2mono(rs[t]);
        cnt += ((u >> bit) == (cand >> bit)) ? 1 : 0;
      }
      for (int o = 16; o; o >>= 1) cnt += __shfl_xor(cnt, o);
      if (cnt >= kneed) prefix = cand; else kneed -= cnt;
    }
    const unsigned uthr = prefix;

    float m = -__builtin_inff();
    for (int t = lane; t < NV_; t += 32) {
      float s = rs[t];
      if (f2mono(s) >= uthr) m = fmaxf(m, s);
    }
    for (int o = 16; o; o >>= 1) m = fmaxf(m, __shfl_xor(m, o));

    float Z = 0.f, S1 = 0.f; int ck = 0;
    for (int t = lane; t < NV_; t += 32) {
      float s = rs[t];
      if (f2mono(s) >= uthr) { float e = __expf(s - m); Z += e; S1 += e * (s - m); ck++; }
    }
    for (int o = 16; o; o >>= 1) { Z += __shfl_xor(Z, o); S1 += __shfl_xor(S1, o); ck += __shfl_xor(ck, o); }
    const float invZ = fast_rcp(Z);

    for (int t = lane; t < NV_; t += 32) {
      float s = rs[t];
      rs[t] = (f2mono(s) >= uthr) ? __expf(s - m) * invZ : 0.f;
    }
    if (lane == 0) {
      float ent = __logf(Z) - S1 * invZ;            // -sum p ln p over kept
      ent += (float)(NV_ - ck) * CLIP * NLCLIP;     // clipped zeros term
      atomicAdd(&scalars[1], ent * (1.0f / (LOGK * (float)(B_ * H_ * NA_))));
      atomicAdd(&scalars[2], ent * (1.0f / (float)(B_ * H_ * NA_)));
    }
  }
  __syncthreads();

  // ---- head-averaged attention accumulation ----
  for (int i = tid; i < 16 * NV_; i += 128) {
    int r = i >> 11, key = i & (NV_ - 1);
    atomicAdd(&attn_avg[(size_t)(b * NA_ + q0 + r) * NV_ + key], sc[r][key] * (1.0f / H_));
  }
  __syncthreads();

  // ---- HV = attn @ Val : Val chunks staged LDS-side via async copy ----
  {
    v8f acc = {};
    const int dcol = wv * 16 + ar;         // this wave's d-columns
    const int vrow = tid >> 2, vq = tid & 3;
    for (int kc = 0; kc < NV_; kc += 32) {
      // async stage 32 keys x 64 d (4 KB), 32B per thread, coalesced
      const bf16_t* src = Vm + (size_t)(b * NV_ + kc + vrow) * DIM_ + h * DH_ + vq * 16;
      async_load_b128(src,     &Vs[vrow][vq * 16]);
      async_load_b128(src + 8, &Vs[vrow][vq * 16 + 8]);
      wait_async0();
      __syncthreads();

      v16bf ap;
      #pragma unroll
      for (int j = 0; j < 8; ++j) {
        ap[j]     = (bf16_t)sc[ar][kc + kb2 + j];
        ap[8 + j] = (bf16_t)sc[ar][kc + kb2 + 8 + j];
      }
      v16bf bv;
      #pragma unroll
      for (int e = 0; e < 16; ++e) bv[e] = Vs[kb2 + e][dcol];
      acc = wmma_bf16(ap, bv, acc);
      __syncthreads();
    }
    const int rr = (lane >> 4) * 8;
    #pragma unroll
    for (int i = 0; i < 8; ++i) {
      int q = q0 + rr + i;
      hvcat[(size_t)(b * NA_ + q) * DIM_ + h * DH_ + dcol] = (bf16_t)acc[i];
    }
  }
}

// ---------------- kernel: attn_max = mean_b,q max_k attn_avg ----------------
__global__ __launch_bounds__(256)
void attnmax_kernel(const float* __restrict__ attn_avg, float* __restrict__ scalars) {
  const int row = blockIdx.x;  // b*NA + q
  const float* r = attn_avg + (size_t)row * NV_;
  float m = -__builtin_inff();
  for (int t = threadIdx.x; t < NV_; t += 256) m = fmaxf(m, r[t]);
  __shared__ float sh[8];
  for (int o = 16; o; o >>= 1) m = fmaxf(m, __shfl_xor(m, o));
  if ((threadIdx.x & 31) == 0) sh[threadIdx.x >> 5] = m;
  __syncthreads();
  if (threadIdx.x == 0) {
    float t = sh[0];
    for (int i = 1; i < 8; ++i) t = fmaxf(t, sh[i]);
    atomicAdd(&scalars[0], t * (1.0f / (float)(B_ * NA_)));
  }
}

// ---------------- host-side launcher ----------------
extern "C" void kernel_launch(void* const* d_in, const int* in_sizes, int n_in,
                              void* d_out, int out_size, void* d_ws, size_t ws_size,
                              hipStream_t stream) {
  const float* A      = (const float*)d_in[0];
  const float* V      = (const float*)d_in[1];
  const unsigned char* Vmask = (const unsigned char*)d_in[2];
  const float* q_w = (const float*)d_in[3];
  const float* q_b = (const float*)d_in[4];
  const float* k_w = (const float*)d_in[5];
  const float* k_b = (const float*)d_in[6];
  const float* v_w = (const float*)d_in[7];
  const float* v_b = (const float*)d_in[8];
  const float* o_w = (const float*)d_in[9];
  const float* o_b = (const float*)d_in[10];
  const float* q_ln_g = (const float*)d_in[11];
  const float* q_ln_b = (const float*)d_in[12];
  const float* v_ln_g = (const float*)d_in[13];
  const float* v_ln_b = (const float*)d_in[14];
  const float* g_ln_g = (const float*)d_in[15];
  const float* g_ln_b = (const float*)d_in[16];
  const float* gate_w = (const float*)d_in[17];
  const float* gate_b = (const float*)d_in[18];

  // output layout (fp32, return-order concat)
  float* out = (float*)d_out;
  float* outHV      = out;                               // [B,NA,DIM]
  float* outHVraw   = out + (size_t)ROWS_A * DIM_;       // [B,NA,DIM]
  float* outScalars = out + (size_t)2 * ROWS_A * DIM_;   // attn_max, ent_norm, ent_abs, gate_mean
  float* outAvg     = outScalars + 4;                    // [B,NA,NV]

  // workspace layout
  char* ws = (char*)d_ws;
  size_t off = 0;
  auto take = [&](size_t bytes) { char* p = ws + off; off += (bytes + 255) & ~(size_t)255; return p; };
  bf16_t* An   = (bf16_t*)take((size_t)ROWS_A * DIM_ * 2);
  bf16_t* Gn   = (bf16_t*)take((size_t)ROWS_A * DIM_ * 2);
  bf16_t* Vn   = (bf16_t*)take((size_t)ROWS_V * DIM_ * 2);
  bf16_t* Qb   = (bf16_t*)take((size_t)ROWS_A * DIM_ * 2);
  bf16_t* Kb   = (bf16_t*)take((size_t)ROWS_V * DIM_ * 2);
  bf16_t* Valb = (bf16_t*)take((size_t)ROWS_V * DIM_ * 2);
  float*  Gm   = (float*) take((size_t)ROWS_A * DIM_ * 4);
  bf16_t* HVc  = (bf16_t*)take((size_t)ROWS_A * DIM_ * 2);
  (void)ws_size; (void)n_in; (void)in_sizes; (void)out_size;

  // 1) zero accumulated outputs (scalars + attn_avg)
  {
    int n = 4 + B_ * NA_ * NV_;
    zero_kernel<<<(n + 255) / 256, 256, 0, stream>>>(outScalars, n);
  }
  // 2) layernorms
  ln_dual_kernel<<<ROWS_A, 256, 0, stream>>>(A, q_ln_g, q_ln_b, g_ln_g, g_ln_b, An, Gn);
  ln_dual_kernel<<<ROWS_V, 256, 0, stream>>>(V, v_ln_g, v_ln_b, v_ln_g, v_ln_b, Vn, nullptr);
  // 3) projections (bf16 WMMA)
  gemm_wmma_kernel<<<dim3(ROWS_A / BM, DIM_ / BN), 128, 0, stream>>>(
      An, q_w, q_b, Qb, nullptr, nullptr, nullptr, nullptr, ROWS_A, 0);
  gemm_wmma_kernel<<<dim3(ROWS_V / BM, DIM_ / BN), 128, 0, stream>>>(
      Vn, k_w, k_b, Kb, nullptr, nullptr, nullptr, nullptr, ROWS_V, 0);
  gemm_wmma_kernel<<<dim3(ROWS_V / BM, DIM_ / BN), 128, 0, stream>>>(
      Vn, v_w, v_b, Valb, nullptr, nullptr, nullptr, nullptr, ROWS_V, 0);
  // 4) attention (scores + top-k softmax + attn@V), one block per (b,h,q-tile)
  attention_kernel<<<B_ * H_ * (NA_ / 16), 128, 0, stream>>>(
      Qb, Kb, Valb, Vmask, HVc, outAvg, outScalars);
  // 5) attn_max reduction over attn_avg
  attnmax_kernel<<<B_ * NA_, 256, 0, stream>>>(outAvg, outScalars);
  // 6) gate projection (sigmoid + gate_mean)
  gemm_wmma_kernel<<<dim3(ROWS_A / BM, DIM_ / BN), 128, 0, stream>>>(
      Gn, gate_w, gate_b, nullptr, Gm, nullptr, nullptr, outScalars + 3, ROWS_A, 1);
  // 7) output projection: HV_raw and gated HV
  gemm_wmma_kernel<<<dim3(ROWS_A / BM, DIM_ / BN), 128, 0, stream>>>(
      HVc, o_w, o_b, nullptr, outHV, Gm, outHVraw, nullptr, ROWS_A, 2);
}
